// GroupedQueryAttention_22308060136146
// MI455X (gfx1250) — matrix operand order repaired and checked by reading, not compiled, not run on hardware
//
#include <hip/hip_runtime.h>
#include <hip/hip_bf16.h>
#include <hip/hip_fp16.h>

// ---------------- problem constants (fixed by reference) ----------------
#define B_    2
#define T_    2048
#define E_    2048
#define HQ_   32
#define HKV_  8
#define HD_   64
#define GROUP_ 4
#define KVE_  512   // HKV*HD

typedef _Float16 h16;
typedef __attribute__((ext_vector_type(16))) _Float16 v16h;
typedef __attribute__((ext_vector_type(8)))  _Float16 v8h;
typedef __attribute__((ext_vector_type(4)))  _Float16 v4h;
typedef __attribute__((ext_vector_type(8)))  float    v8f;
typedef __attribute__((ext_vector_type(4)))  unsigned uint4v;
typedef __attribute__((ext_vector_type(8)))  int      int8v;
typedef __attribute__((ext_vector_type(4)))  int      int4v;

union Frag16 { v16h v; v8h h[2]; unsigned u[8]; };

// D = A(16x32 f16) * B(32x16 f16) + C(16x16 f32)
#define WMMA_F16(A,B,C) \
  __builtin_amdgcn_wmma_f32_16x16x32_f16(false,(A),false,(B),(short)0,(C),false,false)

// ---------------- DPP 16-lane row reductions (VALU only, no LDS) ----------
// xor1: quad_perm(1,0,3,2)=0xB1 ; xor2: quad_perm(2,3,0,1)=0x4E
// xor-within-8: ROW_HALF_MIRROR=0x141 ; xor-within-16: ROW_MIRROR=0x140
template<int CTRL>
__device__ __forceinline__ float dppf(float x) {
  return __int_as_float(
      __builtin_amdgcn_update_dpp(0, __float_as_int(x), CTRL, 0xF, 0xF, true));
}
__device__ __forceinline__ float rowmax16(float v) {
  v = fmaxf(v, dppf<0xB1>(v));
  v = fmaxf(v, dppf<0x4E>(v));
  v = fmaxf(v, dppf<0x141>(v));
  v = fmaxf(v, dppf<0x140>(v));
  return v;
}
__device__ __forceinline__ float rowsum16(float v) {
  v += dppf<0xB1>(v);
  v += dppf<0x4E>(v);
  v += dppf<0x141>(v);
  v += dppf<0x140>(v);
  return v;
}

// ---------------- TDM: 2D tile (tile_w elems x tile_h rows) -> LDS --------
// data_size = 2B. LDS rows padded: +4 dwords after every 32 dwords (row = 64
// halves -> stored row stride = 72 halves). group0/group1 per ISA 8.3/8.4.
// This toolchain uses the 6-arg builtin: (g0, g1, g2, g3, g_extra, cpol).
__device__ __forceinline__ void tdm_load_2d(unsigned lds_byte_addr,
                                            const h16* gaddr,
                                            unsigned tile_w, unsigned tile_h,
                                            unsigned stride_elems) {
  unsigned long long ga = (unsigned long long)(size_t)gaddr;
  uint4v g0;
  g0[0] = 1u;                                        // count=1, user mode
  g0[1] = lds_byte_addr;                             // lds_addr (bytes)
  g0[2] = (unsigned)(ga & 0xFFFFFFFFu);              // global_addr[31:0]
  g0[3] = (unsigned)((ga >> 32) & 0x01FFFFFFu) | (2u << 30);  // addr[56:32]|type=2
  int8v g1;
  g1[0] = (int)((1u << 16) |            // data_size = 2B
                (1u << 20) |            // pad_enable
                (4u << 22) |            // pad_interval: 32 dwords
                (3u << 25));            // pad_amount: 4 dwords
  g1[1] = (int)(tile_w << 16);          // tensor_dim0[15:0]
  g1[2] = (int)(tile_h << 16);          // tensor_dim1[15:0]
  g1[3] = (int)(tile_w << 16);          // tile_dim0
  g1[4] = (int)(tile_h);                // tile_dim1 (tile_dim2 = 0 -> 2D)
  g1[5] = (int)stride_elems;            // tensor_dim0_stride lo32
  g1[6] = 0;
  g1[7] = 0;
  int4v z4 = {0, 0, 0, 0};
  int8v z8 = {0, 0, 0, 0, 0, 0, 0, 0};
  __builtin_amdgcn_tensor_load_to_lds(g0, g1, z4, z4, z8, 0);
}

// ---------------- f32 -> f16 cast ----------------
__global__ void cast_f32_to_f16(const float* __restrict__ s, h16* __restrict__ d, int n) {
  int i = (blockIdx.x * blockDim.x + threadIdx.x) * 4;
  int stride = gridDim.x * blockDim.x * 4;
  for (; i < n; i += stride) {
    float4 f = *(const float4*)(s + i);
    v4h o;
    o[0] = (h16)f.x; o[1] = (h16)f.y; o[2] = (h16)f.z; o[3] = (h16)f.w;
    *(v4h*)(d + i) = o;
  }
}

// ---------------- tiled WMMA GEMM: C[M,N] = A[M,K] @ B[K,N] ----------------
// 256 threads = 8 waves. Tile BM=128, BN=128, BK=32. Wave (wr,wc) in 4x2 grid
// owns 32 rows x 64 cols = 8 WMMA tiles per k-step.
template<bool F16OUT>
__global__ void gemm_wmma(const h16* __restrict__ A, const h16* __restrict__ Bm,
                          h16* __restrict__ Cf16, float* __restrict__ Cf32,
                          const float* __restrict__ bias, float scale,
                          int M, int N, int K) {
  constexpr int BM = 128, BN = 128, BK = 32;
  constexpr int LDA = BK + 8;    // 40 halves (16B-aligned rows)
  constexpr int LDB = BN + 4;    // 132 halves
  __shared__ h16 As[BM][LDA];
  __shared__ h16 Bs[BK][LDB];

  const int tid  = threadIdx.x;
  const int wave = tid >> 5;
  const int lane = tid & 31;
  const int l15  = lane & 15;
  const int l16  = lane >> 4;
  const int wr   = wave & 3;     // row block (32 rows)
  const int wc   = wave >> 2;    // col block (64 cols)

  const int m0 = blockIdx.y * BM;
  const int n0 = blockIdx.x * BN;

  v8f acc[2][4] = {};

  const int arow = tid >> 1;             // 0..127
  const int acol = (tid & 1) * 16;       // 0/16
  const int brow = tid >> 3;             // 0..31
  const int bcol = (tid & 7) * 16;       // 0..112

  for (int k0 = 0; k0 < K; k0 += BK) {
    Frag16 ta, tb;
    ta.v = *(const v16h*)(A  + (size_t)(m0 + arow) * K + k0 + acol);
    tb.v = *(const v16h*)(Bm + (size_t)(k0 + brow) * N + n0 + bcol);
    *(v8h*)&As[arow][acol]     = ta.h[0];
    *(v8h*)&As[arow][acol + 8] = ta.h[1];
    {
      unsigned* bp = (unsigned*)&Bs[brow][bcol];
#pragma unroll
      for (int i = 0; i < 8; ++i) bp[i] = tb.u[i];
    }
    __syncthreads();

    Frag16 af[2], bf[4];
#pragma unroll
    for (int a = 0; a < 2; ++a) {
      const h16* ap = &As[wr * 32 + a * 16 + l15][0];
      af[a].h[0] = *(const v8h*)(ap + l16 * 8);
      af[a].h[1] = *(const v8h*)(ap + l16 * 8 + 16);
    }
#pragma unroll
    for (int t = 0; t < 4; ++t) {
#pragma unroll
      for (int h = 0; h < 16; ++h)
        bf[t].v[h] = Bs[l16 * 8 + (h & 7) + (h >> 3) * 16][wc * 64 + t * 16 + l15];
    }
#pragma unroll
    for (int a = 0; a < 2; ++a)
#pragma unroll
      for (int t = 0; t < 4; ++t)
        acc[a][t] = WMMA_F16(af[a].v, bf[t].v, acc[a][t]);
    __syncthreads();
  }

#pragma unroll
  for (int a = 0; a < 2; ++a) {
#pragma unroll
    for (int t = 0; t < 4; ++t) {
#pragma unroll
      for (int r = 0; r < 8; ++r) {
        const int m = m0 + wr * 32 + a * 16 + l16 * 8 + r;
        const int n = n0 + wc * 64 + t * 16 + l15;
        const float val = acc[a][t][r];
        if constexpr (F16OUT) Cf16[(size_t)m * N + n] = (h16)(val * scale);
        else                  Cf32[(size_t)m * N + n] = val + bias[n];
      }
    }
  }
}

// ---------------- flash attention ----------------
// grid: B*HKV*(T/16) blocks, 128 threads (4 waves = 4 grouped query heads).
// K/V blocks staged into LDS by the Tensor Data Mover (wave 0 issues D#).
__global__ void attn_wmma(const h16* __restrict__ Q, const h16* __restrict__ Kk,
                          const h16* __restrict__ Vv, h16* __restrict__ Ctx) {
  constexpr int TK  = 32;          // keys per iteration
  constexpr int LDK = HD_ + 8;     // 72 halves = TDM padded row stride
  __shared__ h16 Ks[TK][LDK];
  __shared__ h16 Vs[TK][LDK];
  __shared__ h16 Ps[4][16][40];    // per-wave P scratch (16 q x 32 k)

  const int tid  = threadIdx.x;
  const int w    = tid >> 5;       // group head index g
  const int lane = tid & 31;
  const int l15  = lane & 15;
  const int l16  = lane >> 4;

  const int qt  = blockIdx.x % (T_ / 16);
  const int hkv = (blockIdx.x / (T_ / 16)) % HKV_;
  const int b   = blockIdx.x / ((T_ / 16) * HKV_);
  const int hq  = hkv * GROUP_ + w;

  // Q fragments (A layout), d = 0..31 and 32..63 (Q pre-scaled by 1/8)
  const h16* qbase = Q + (size_t)(b * T_ + qt * 16 + l15) * E_ + hq * HD_;
  Frag16 qa0, qa1;
  qa0.h[0] = *(const v8h*)(qbase + l16 * 16);
  qa0.h[1] = *(const v8h*)(qbase + l16 * 16 + 8);
  qa1.h[0] = *(const v8h*)(qbase + 32 + l16 * 16);
  qa1.h[1] = *(const v8h*)(qbase + 32 + l16 * 16 + 8);

  v8f o[4] = {};
  float mrow[8], lrow[8];
#pragma unroll
  for (int r = 0; r < 8; ++r) { mrow[r] = -1e30f; lrow[r] = 0.0f; }

  const h16* Kp = Kk + (size_t)b * T_ * KVE_ + hkv * HD_;
  const h16* Vp = Vv + (size_t)b * T_ * KVE_ + hkv * HD_;
  const unsigned ksBase = (unsigned)(size_t)&Ks[0][0];
  const unsigned vsBase = (unsigned)(size_t)&Vs[0][0];

  for (int kb = 0; kb < T_; kb += TK) {
    // ---- TDM stage: K/V 32x64 tiles, row stride 512 elems, padded rows ----
    if (w == 0) {
      tdm_load_2d(ksBase, Kp + (size_t)kb * KVE_, HD_, TK, KVE_);
      tdm_load_2d(vsBase, Vp + (size_t)kb * KVE_, HD_, TK, KVE_);
      __builtin_amdgcn_s_wait_tensorcnt(0);
    }
    __syncthreads();

    // S = Q @ K^T : two 16x16 tiles (keys 0..15 / 16..31), K-dim = HD = 2x32
    v8f s0 = {}, s1 = {};
    {
      Frag16 kf0, kf1;                 // col = key l15 ; k-dim d = g*16 + h
      kf0.h[0] = *(const v8h*)(&Ks[l15][l16 * 16]);
      kf0.h[1] = *(const v8h*)(&Ks[l15][l16 * 16 + 8]);
      kf1.h[0] = *(const v8h*)(&Ks[l15][32 + l16 * 16]);
      kf1.h[1] = *(const v8h*)(&Ks[l15][32 + l16 * 16 + 8]);
      s0 = WMMA_F16(qa0.v, kf0.v, s0);
      s0 = WMMA_F16(qa1.v, kf1.v, s0);
      kf0.h[0] = *(const v8h*)(&Ks[16 + l15][l16 * 16]);
      kf0.h[1] = *(const v8h*)(&Ks[16 + l15][l16 * 16 + 8]);
      kf1.h[0] = *(const v8h*)(&Ks[16 + l15][32 + l16 * 16]);
      kf1.h[1] = *(const v8h*)(&Ks[16 + l15][32 + l16 * 16 + 8]);
      s1 = WMMA_F16(qa0.v, kf0.v, s1);
      s1 = WMMA_F16(qa1.v, kf1.v, s1);
    }

    // online softmax per row (DPP reductions, no LDS)
    float corr[8];
#pragma unroll
    for (int r = 0; r < 8; ++r) {
      const float tmax = rowmax16(fmaxf(s0[r], s1[r]));
      const float mnew = fmaxf(mrow[r], tmax);
      const float c    = __expf(mrow[r] - mnew);
      const float p0   = __expf(s0[r] - mnew);
      const float p1   = __expf(s1[r] - mnew);
      const float rs   = rowsum16(p0 + p1);
      lrow[r] = lrow[r] * c + rs;
      mrow[r] = mnew;
      corr[r] = c;
      Ps[w][l16 * 8 + r][l15]      = (h16)p0;   // C-layout -> row-major P
      Ps[w][l16 * 8 + r][16 + l15] = (h16)p1;
    }
#pragma unroll
    for (int r = 0; r < 8; ++r) {
      o[0][r] *= corr[r]; o[1][r] *= corr[r];
      o[2][r] *= corr[r]; o[3][r] *= corr[r];
    }
    __syncthreads();

    // P fragment (A layout) from LDS scratch
    Frag16 pf;
    const h16* pp = &Ps[w][l15][0];
    pf.h[0] = *(const v8h*)(pp + l16 * 16);
    pf.h[1] = *(const v8h*)(pp + l16 * 16 + 8);

    // O += P(16x32) @ V(32x64): 4 d-tiles
#pragma unroll
    for (int t = 0; t < 4; ++t) {
      Frag16 vf;                      // col = d t*16+l15 ; k = key g*16 + h
#pragma unroll
      for (int h = 0; h < 16; ++h) vf.v[h] = Vs[l16 * 16 + h][t * 16 + l15];
      o[t] = WMMA_F16(pf.v, vf.v, o[t]);
    }
    __syncthreads();
  }

  // normalize and store ctx[b, t, hq*HD + d] as f16
#pragma unroll
  for (int t = 0; t < 4; ++t) {
#pragma unroll
    for (int r = 0; r < 8; ++r) {
      const int tq = qt * 16 + l16 * 8 + r;
      const float val = o[t][r] / lrow[r];
      Ctx[(size_t)(b * T_ + tq) * E_ + hq * HD_ + t * 16 + l15] = (h16)val;
    }
  }
}

// ---------------- launcher ----------------
extern "C" void kernel_launch(void* const* d_in, const int* in_sizes, int n_in,
                              void* d_out, int out_size, void* d_ws, size_t ws_size,
                              hipStream_t stream) {
  (void)in_sizes; (void)n_in; (void)out_size; (void)ws_size;
  const float* x  = (const float*)d_in[0];
  const float* Wq = (const float*)d_in[1];
  const float* Wk = (const float*)d_in[2];
  const float* Wv = (const float*)d_in[3];
  const float* Wo = (const float*)d_in[4];
  const float* bo = (const float*)d_in[5];
  float* out = (float*)d_out;

  char* ws = (char*)d_ws;
  size_t off = 0;
  auto alloc = [&](size_t nelem) -> h16* {
    h16* p = (h16*)(ws + off); off += nelem * sizeof(h16); return p;
  };
  const size_t NX  = (size_t)B_ * T_ * E_;
  const size_t NWQ = (size_t)E_ * E_;
  const size_t NWK = (size_t)E_ * KVE_;
  const size_t NKV = (size_t)B_ * T_ * KVE_;

  h16* x_h  = alloc(NX);
  h16* Wq_h = alloc(NWQ);
  h16* Wk_h = alloc(NWK);
  h16* Wv_h = alloc(NWK);
  h16* Wo_h = alloc(NWQ);
  h16* Q_h  = alloc(NX);
  h16* K_h  = alloc(NKV);
  h16* V_h  = alloc(NKV);
  h16* C_h  = alloc(NX);

  const int ct = 256;
  auto cblk = [&](size_t n) { return (int)((n / 4 + ct - 1) / ct); };
  cast_f32_to_f16<<<cblk(NX),  ct, 0, stream>>>(x,  x_h,  (int)NX);
  cast_f32_to_f16<<<cblk(NWQ), ct, 0, stream>>>(Wq, Wq_h, (int)NWQ);
  cast_f32_to_f16<<<cblk(NWK), ct, 0, stream>>>(Wk, Wk_h, (int)NWK);
  cast_f32_to_f16<<<cblk(NWK), ct, 0, stream>>>(Wv, Wv_h, (int)NWK);
  cast_f32_to_f16<<<cblk(NWQ), ct, 0, stream>>>(Wo, Wo_h, (int)NWQ);

  const dim3 blk(256);
  const int M = B_ * T_;                               // 4096
  gemm_wmma<true><<<dim3(E_ / 128,   M / 128), blk, 0, stream>>>(
      x_h, Wq_h, Q_h, nullptr, nullptr, 0.125f, M, E_,  E_);
  gemm_wmma<true><<<dim3(KVE_ / 128, M / 128), blk, 0, stream>>>(
      x_h, Wk_h, K_h, nullptr, nullptr, 1.0f,   M, KVE_, E_);
  gemm_wmma<true><<<dim3(KVE_ / 128, M / 128), blk, 0, stream>>>(
      x_h, Wv_h, V_h, nullptr, nullptr, 1.0f,   M, KVE_, E_);

  attn_wmma<<<B_ * HKV_ * (T_ / 16), dim3(128), 0, stream>>>(Q_h, K_h, V_h, C_h);

  gemm_wmma<false><<<dim3(E_ / 128, M / 128), blk, 0, stream>>>(
      C_h, Wo_h, nullptr, out, bo, 1.0f, M, E_, E_);
}
